// LatentAttention_71416716198045
// MI455X (gfx1250) — compile-verified, matrix operand order checked by reading, not run on hardware
//
#include <hip/hip_runtime.h>
#include <stdint.h>

// ---------------------------------------------------------------------------
// MLA latent attention for gfx1250 (MI455X).
// bf16 WMMA pipeline + async global->LDS double buffering (ASYNCcnt).
// ---------------------------------------------------------------------------

typedef __attribute__((ext_vector_type(16))) __bf16 v16bf;
typedef __attribute__((ext_vector_type(8)))  __bf16 v8bf;
typedef __attribute__((ext_vector_type(8)))  float  v8f;

#define DEVFN __device__ __forceinline__

constexpr int Bn = 2;
constexpr int Tn = 2048;
constexpr int Hn = 16;
constexpr int Ln = 64;
constexpr int Cn = 1024;            // n_embd == n_head * latent
constexpr float ATT_SCALE = 0.125f; // 1/sqrt(64)
constexpr int SP = 72;              // padded LDS row stride (elems) for 64-wide tiles

// ---------------------------------------------------------------------------
// CDNA5 async global->LDS copy (16B per lane), tracked by ASYNCcnt.
// ---------------------------------------------------------------------------
DEVFN void async_ld16(__bf16* lds, const __bf16* g) {
  const uint32_t loff = (uint32_t)(uintptr_t)lds;   // low 32 bits = LDS offset
  asm volatile("global_load_async_to_lds_b128 %0, %1, off"
               :: "v"(loff), "v"((uint64_t)(uintptr_t)g)
               : "memory");
}
DEVFN void wait_async0() {
  asm volatile("s_wait_asynccnt 0" ::: "memory");
}

// ---------------------------------------------------------------------------
// Fragment helpers (wave32 WMMA 16-bit layouts).
// A 16x32 frag: lane holds row M=lane&15; element e -> K = kb + e + (e>=8)*8,
// kb = 0 (lanes 0-15) or 8 (lanes 16-31).  B^T stored row-major [N,K] loads
// identically with N in place of M.  Works for global or LDS pointers.
// ---------------------------------------------------------------------------
DEVFN v16bf load_frag(const __bf16* __restrict__ base, int r0, int k0, int ld, int lane) {
  const int r  = r0 + (lane & 15);
  const int kb = k0 + ((lane >> 4) << 3);
  const __bf16* p = base + (size_t)r * ld + kb;
  v8bf lo = *(const v8bf*)p;          // K = kb .. kb+7
  v8bf hi = *(const v8bf*)(p + 16);   // K = kb+16 .. kb+23
  v16bf f;
#pragma unroll
  for (int i = 0; i < 8; ++i) { f[i] = lo[i]; f[i + 8] = hi[i]; }
  return f;
}

DEVFN v8f wmma_bf16(v16bf a, v16bf b, v8f c) {
  return __builtin_amdgcn_wmma_f32_16x16x32_bf16(false, a, false, b, (short)0, c,
                                                 false, false);
}

// ---------------------------------------------------------------------------
// Tile stagers (async).
// ---------------------------------------------------------------------------
DEVFN void stage128x64(const __bf16* __restrict__ g, int row0, int k0, int ld,
                       __bf16* lbuf, int tid) {
#pragma unroll
  for (int c = 0; c < 4; ++c) {
    const int chunk = tid * 4 + c;       // 0..1023
    const int row = chunk >> 3;          // 0..127
    const int kc  = (chunk & 7) << 3;    // 0..56 step 8
    async_ld16(lbuf + row * SP + kc,
               g + (size_t)(row0 + row) * ld + k0 + kc);
  }
}

DEVFN void stage256x64(const __bf16* __restrict__ g, int row0, int k0, int ld,
                       __bf16* lbuf, int tid) {
#pragma unroll
  for (int c = 0; c < 8; ++c) {
    const int chunk = tid * 8 + c;       // 0..2047
    const int row = chunk >> 3;          // 0..255
    const int kc  = (chunk & 7) << 3;
    async_ld16(lbuf + row * SP + kc,
               g + (size_t)(row0 + row) * ld + k0 + kc);
  }
}

DEVFN void stage64x64(const __bf16* __restrict__ g, int row0, int col0, int ld,
                      __bf16* lbuf, int tid) {
#pragma unroll
  for (int c = 0; c < 4; ++c) {
    const int chunk = tid * 4 + c;       // 0..511
    const int row = chunk >> 3;          // 0..63
    const int kc  = (chunk & 7) << 3;
    async_ld16(lbuf + row * SP + kc,
               g + (size_t)(row0 + row) * ld + col0 + kc);
  }
}

// ---------------------------------------------------------------------------
// Prep kernels: f32 -> bf16 conversion / transpose / Wq*Wc folding.
// ---------------------------------------------------------------------------
__global__ void f32_to_bf16_kernel(const float* __restrict__ in,
                                   __bf16* __restrict__ out, int n) {
  int i = blockIdx.x * blockDim.x + threadIdx.x;
  if (i < n) out[i] = (__bf16)in[i];
}

// WT[n][k] = W[k][n]  (bf16)
__global__ void transpose_to_bf16_kernel(const float* __restrict__ W,
                                         __bf16* __restrict__ WT, int K, int N) {
  int i = blockIdx.x * blockDim.x + threadIdx.x;
  if (i >= K * N) return;
  int nI = i % N;
  int kI = i / N;
  WT[(size_t)nI * K + kI] = (__bf16)W[(size_t)kI * N + nI];
}

// WqcT[j][c] = sum_d Wq[c][h*64+d] * Wc[d][l]   with j = h*64 + l
__global__ void make_wqcT_kernel(const float* __restrict__ Wq,
                                 const float* __restrict__ Wc,
                                 __bf16* __restrict__ WqcT) {
  int i = blockIdx.x * blockDim.x + threadIdx.x;
  if (i >= Cn * Cn) return;
  int j = i / Cn;
  int c = i % Cn;
  int h = j >> 6, l = j & 63;
  const float* wq = Wq + (size_t)c * Cn + h * 64;
  float acc = 0.f;
#pragma unroll 8
  for (int d = 0; d < 64; ++d) acc += wq[d] * Wc[d * 64 + l];
  WqcT[(size_t)j * Cn + c] = (__bf16)acc;
}

__global__ void make_bqc_kernel(const float* __restrict__ bq,
                                const float* __restrict__ Wc,
                                float* __restrict__ bqc) {
  int j = blockIdx.x * blockDim.x + threadIdx.x;
  if (j >= Cn) return;
  int h = j >> 6, l = j & 63;
  float acc = 0.f;
  for (int d = 0; d < 64; ++d) acc += bq[h * 64 + d] * Wc[d * 64 + l];
  bqc[j] = acc;
}

// ---------------------------------------------------------------------------
// GEMM: C[M,N] = A[M,K](bf16) * BT[N,K]^T(bf16) + bias.
// Block = 256 threads (8 waves), 128x256 tile, K-step 64, double-buffered
// async LDS staging. Wave = 64x64 (4x4 frags) -> ds:wmma ratio 1:1.
// MODE 0: f32 row-major [M,N]
// MODE 1: bf16 [B,H,T,L]   (row=b*T+t, col=h*64+l)
// MODE 2: bf16 [B,H,L,T]   (per-head transposed, V as future B-operand)
// ---------------------------------------------------------------------------
template <int MODE>
__global__ void __launch_bounds__(256)
gemm_wmma_kernel(const __bf16* __restrict__ A, const __bf16* __restrict__ BT,
                 const float* __restrict__ bias, void* __restrict__ outp,
                 int M, int N, int K) {
  __shared__ __align__(16) __bf16 sA[2][128 * SP];
  __shared__ __align__(16) __bf16 sB[2][256 * SP];

  const int tid  = threadIdx.x;
  const int lane = tid & 31;
  const int wave = tid >> 5;
  const int wm = wave >> 2, wn = wave & 3;
  const int bm = blockIdx.x * 128;
  const int bn = blockIdx.y * 256;

  stage128x64(A,  bm, 0, K, sA[0], tid);
  stage256x64(BT, bn, 0, K, sB[0], tid);

  v8f acc[4][4] = {};
  const int nsteps = K >> 6;             // K/64
  for (int s = 0; s < nsteps; ++s) {
    const int b = s & 1;
    wait_async0();
    __syncthreads();                     // tile b complete on all waves
    if (s + 1 < nsteps) {                // prefetch next tile into b^1
      stage128x64(A,  bm, (s + 1) << 6, K, sA[b ^ 1], tid);
      stage256x64(BT, bn, (s + 1) << 6, K, sB[b ^ 1], tid);
    }
#pragma unroll
    for (int kk = 0; kk < 64; kk += 32) {
      v16bf a[4], bb[4];
#pragma unroll
      for (int i = 0; i < 4; ++i)
        a[i] = load_frag(sA[b], wm * 64 + i * 16, kk, SP, lane);
#pragma unroll
      for (int j = 0; j < 4; ++j)
        bb[j] = load_frag(sB[b], wn * 64 + j * 16, kk, SP, lane);
#pragma unroll
      for (int i = 0; i < 4; ++i)
#pragma unroll
        for (int j = 0; j < 4; ++j)
          acc[i][j] = wmma_bf16(a[i], bb[j], acc[i][j]);
    }
  }

  const int nlo = lane & 15;
  const int rhi = (lane >> 4) << 3;
#pragma unroll
  for (int i = 0; i < 4; ++i) {
#pragma unroll
    for (int j = 0; j < 4; ++j) {
      const int col = bn + wn * 64 + j * 16 + nlo;
      const float bbv = bias[col];
#pragma unroll
      for (int e = 0; e < 8; ++e) {
        const int row = bm + wm * 64 + i * 16 + e + rhi;
        const float v = acc[i][j][e] + bbv;
        if (MODE == 0) {
          ((float*)outp)[(size_t)row * N + col] = v;
        } else {
          const int bI = row >> 11, t = row & (Tn - 1);
          const int h = col >> 6, l = col & 63;
          size_t idx;
          if (MODE == 1) idx = (((size_t)(bI * Hn + h) * Tn) + t) * Ln + l;
          else           idx = (((size_t)(bI * Hn + h) * Ln) + l) * Tn + t;
          ((__bf16*)outp)[idx] = (__bf16)v;
        }
      }
    }
  }
}

// ---------------------------------------------------------------------------
// Flash attention: grid = (T/64, B*H), block = 128 (4 waves x 16 query rows).
// qc,k: [B,H,T,L] bf16; vT: [B,H,L,T] bf16; y: [B,T,C] bf16.
// K/V^T tiles double-buffered in LDS via async loads and shared by all waves;
// causal masking is branchless so all waves hit every barrier.
// ---------------------------------------------------------------------------
__global__ void __launch_bounds__(128)
mla_attn_kernel(const __bf16* __restrict__ qc, const __bf16* __restrict__ kk,
                const __bf16* __restrict__ vT, __bf16* __restrict__ y) {
  __shared__ __align__(16) __bf16 sK[2][64 * SP];
  __shared__ __align__(16) __bf16 sV[2][64 * SP];
  __shared__ __align__(16) __bf16 sP[4][16 * SP];

  const int tid  = threadIdx.x;
  const int lane = tid & 31;
  const int wave = tid >> 5;             // 0..3
  const int bh = blockIdx.y;             // b*16 + h
  const int bI = bh >> 4, h = bh & 15;
  const int r0 = blockIdx.x * 64 + wave * 16;

  const __bf16* qch = qc + (size_t)bh * Tn * Ln;
  const __bf16* kh  = kk + (size_t)bh * Tn * Ln;
  const __bf16* vh  = vT + (size_t)bh * Ln * Tn;
  __bf16* pp = sP[wave];                 // per-wave P scratch

  const v16bf qa0 = load_frag(qch, r0, 0,  Ln, lane);
  const v16bf qa1 = load_frag(qch, r0, 32, Ln, lane);

  v8f o[4] = {};
  float mrow[8], lrow[8];
#pragma unroll
  for (int j = 0; j < 8; ++j) { mrow[j] = -3.0e38f; lrow[j] = 0.f; }

  const int nlo = lane & 15;
  const int rhi = (lane >> 4) << 3;
  const int nblk = blockIdx.x + 1;       // causal: s-blocks 0..rb

  stage64x64(kh, 0, 0, Ln, sK[0], tid);
  stage64x64(vh, 0, 0, Tn, sV[0], tid);

  for (int blk = 0; blk < nblk; ++blk) {
    const int s0 = blk * 64;
    const int b = blk & 1;
    wait_async0();
    __syncthreads();
    if (blk + 1 < nblk) {
      stage64x64(kh, s0 + 64, 0, Ln, sK[b ^ 1], tid);
      stage64x64(vh, 0, s0 + 64, Tn, sV[b ^ 1], tid);
    }

    // ---- scores S = qc @ k^T : 16x64 in 4 frags ----
    v8f sf[4];
#pragma unroll
    for (int n = 0; n < 4; ++n) {
      v16bf b0 = load_frag(sK[b], n * 16, 0,  SP, lane);
      v16bf b1 = load_frag(sK[b], n * 16, 32, SP, lane);
      v8f z = {};
      z = wmma_bf16(qa0, b0, z);
      z = wmma_bf16(qa1, b1, z);
      sf[n] = z;
    }

    // ---- scale + causal mask (branchless; masked rows give p == 0) ----
#pragma unroll
    for (int n = 0; n < 4; ++n) {
      const int cs = s0 + n * 16 + nlo;
#pragma unroll
      for (int j = 0; j < 8; ++j) {
        const int row = r0 + j + rhi;
        const float sv = sf[n][j] * ATT_SCALE;
        sf[n][j] = (cs > row) ? -3.0e38f : sv;
      }
    }

    // ---- row max over 64 columns (16-lane shfl groups) ----
    float rmax[8];
#pragma unroll
    for (int j = 0; j < 8; ++j)
      rmax[j] = fmaxf(fmaxf(sf[0][j], sf[1][j]), fmaxf(sf[2][j], sf[3][j]));
#pragma unroll
    for (int j = 0; j < 8; ++j) {
      rmax[j] = fmaxf(rmax[j], __shfl_xor(rmax[j], 1, 32));
      rmax[j] = fmaxf(rmax[j], __shfl_xor(rmax[j], 2, 32));
      rmax[j] = fmaxf(rmax[j], __shfl_xor(rmax[j], 4, 32));
      rmax[j] = fmaxf(rmax[j], __shfl_xor(rmax[j], 8, 32));
    }

    // ---- online softmax rescale ----
    float rs[8];
#pragma unroll
    for (int j = 0; j < 8; ++j) {
      const float mn = fmaxf(mrow[j], rmax[j]);
      const float corr = __expf(mrow[j] - mn);
      mrow[j] = mn;
      lrow[j] *= corr;
#pragma unroll
      for (int n = 0; n < 4; ++n) o[n][j] *= corr;
      rs[j] = 0.f;
    }
#pragma unroll
    for (int n = 0; n < 4; ++n)
#pragma unroll
      for (int j = 0; j < 8; ++j) {
        const float p = __expf(sf[n][j] - mrow[j]);
        sf[n][j] = p;
        rs[j] += p;
      }
#pragma unroll
    for (int j = 0; j < 8; ++j) {
      rs[j] += __shfl_xor(rs[j], 1, 32);
      rs[j] += __shfl_xor(rs[j], 2, 32);
      rs[j] += __shfl_xor(rs[j], 4, 32);
      rs[j] += __shfl_xor(rs[j], 8, 32);
      lrow[j] += rs[j];
    }

    // ---- P -> LDS (bf16, row-major 16xSP), reload as A-frags ----
#pragma unroll
    for (int n = 0; n < 4; ++n)
#pragma unroll
      for (int j = 0; j < 8; ++j)
        pp[(j + rhi) * SP + n * 16 + nlo] = (__bf16)sf[n][j];

    v16bf pa0 = load_frag(pp, 0, 0,  SP, lane);
    v16bf pa1 = load_frag(pp, 0, 32, SP, lane);

    // ---- O += P @ V  (V^T tile in LDS is the B-operand) ----
#pragma unroll
    for (int n = 0; n < 4; ++n) {
      v16bf vb0 = load_frag(sV[b], n * 16, 0,  SP, lane);
      v16bf vb1 = load_frag(sV[b], n * 16, 32, SP, lane);
      o[n] = wmma_bf16(pa0, vb0, o[n]);
      o[n] = wmma_bf16(pa1, vb1, o[n]);
    }
  }

  // ---- normalize + write y[B,T,C] bf16 ----
#pragma unroll
  for (int j = 0; j < 8; ++j) {
    const float inv = 1.0f / lrow[j];
    const int t = r0 + j + rhi;
#pragma unroll
    for (int n = 0; n < 4; ++n) {
      const size_t idx = ((size_t)(bI * Tn + t)) * Cn + h * 64 + n * 16 + nlo;
      y[idx] = (__bf16)(o[n][j] * inv);
    }
  }
}

// ---------------------------------------------------------------------------
// Host launcher
// ---------------------------------------------------------------------------
extern "C" void kernel_launch(void* const* d_in, const int* in_sizes, int n_in,
                              void* d_out, int out_size, void* d_ws, size_t ws_size,
                              hipStream_t stream) {
  (void)in_sizes; (void)n_in; (void)out_size; (void)ws_size;
  const float* x  = (const float*)d_in[0];
  const float* Wq = (const float*)d_in[1];
  const float* bq = (const float*)d_in[2];
  const float* Wk = (const float*)d_in[3];
  const float* bk = (const float*)d_in[4];
  const float* Wv = (const float*)d_in[5];
  const float* bv = (const float*)d_in[6];
  const float* Wo = (const float*)d_in[7];
  const float* bo = (const float*)d_in[8];
  const float* Wc = (const float*)d_in[9];
  float* out = (float*)d_out;

  char* ws = (char*)d_ws;
  size_t off = 0;
  auto alloc = [&](size_t bytes) -> void* {
    void* p = (void*)(ws + off);
    off += (bytes + 255) & ~(size_t)255;
    return p;
  };

  const int M = Bn * Tn;                                     // 4096
  __bf16* xb   = (__bf16*)alloc((size_t)M * Cn * 2);         // x in bf16
  __bf16* WqcT = (__bf16*)alloc((size_t)Cn * Cn * 2);        // (Wq*Wc)^T
  __bf16* WkT  = (__bf16*)alloc((size_t)Cn * Cn * 2);
  __bf16* WvT  = (__bf16*)alloc((size_t)Cn * Cn * 2);
  __bf16* WoT  = (__bf16*)alloc((size_t)Cn * Cn * 2);
  float*  bqc  = (float*)alloc((size_t)Cn * 4);
  __bf16* qcb  = (__bf16*)alloc((size_t)M * Cn * 2);         // [B,H,T,L]
  __bf16* kb   = (__bf16*)alloc((size_t)M * Cn * 2);         // [B,H,T,L]
  __bf16* vTb  = (__bf16*)alloc((size_t)M * Cn * 2);         // [B,H,L,T]
  __bf16* yb   = (__bf16*)alloc((size_t)M * Cn * 2);         // [B,T,C]

  const int nElem = M * Cn;
  f32_to_bf16_kernel<<<(nElem + 255) / 256, 256, 0, stream>>>(x, xb, nElem);
  make_wqcT_kernel<<<(Cn * Cn + 255) / 256, 256, 0, stream>>>(Wq, Wc, WqcT);
  make_bqc_kernel<<<(Cn + 255) / 256, 256, 0, stream>>>(bq, Wc, bqc);
  transpose_to_bf16_kernel<<<(Cn * Cn + 255) / 256, 256, 0, stream>>>(Wk, WkT, Cn, Cn);
  transpose_to_bf16_kernel<<<(Cn * Cn + 255) / 256, 256, 0, stream>>>(Wv, WvT, Cn, Cn);
  transpose_to_bf16_kernel<<<(Cn * Cn + 255) / 256, 256, 0, stream>>>(Wo, WoT, Cn, Cn);

  dim3 gemm_grid(M / 128, Cn / 256);
  gemm_wmma_kernel<1><<<gemm_grid, 256, 0, stream>>>(xb, WqcT, bqc, qcb, M, Cn, Cn);
  gemm_wmma_kernel<1><<<gemm_grid, 256, 0, stream>>>(xb, WkT,  bk,  kb,  M, Cn, Cn);
  gemm_wmma_kernel<2><<<gemm_grid, 256, 0, stream>>>(xb, WvT,  bv,  vTb, M, Cn, Cn);

  mla_attn_kernel<<<dim3(Tn / 64, Bn * Hn), 128, 0, stream>>>(qcb, kb, vTb, yb);

  gemm_wmma_kernel<0><<<gemm_grid, 256, 0, stream>>>(yb, WoT, bo, out, M, Cn, Cn);
}
